// GAT_80118319940093
// MI455X (gfx1250) — matrix operand order repaired and checked by reading, not compiled, not run on hardware
//
#include <hip/hip_runtime.h>
#include <hip/hip_fp16.h>
#include <math.h>

// ---------------- problem constants (match reference) ----------------
#define N_NODES   100000
#define N_EDGES   1600000
#define ETOT      (N_EDGES + N_NODES)   // edges + self loops
#define F_IN      128
#define HEADS1    8
#define C1        16
#define NGRAPH    64
#define NEG_SLOPE 0.2f

typedef __attribute__((ext_vector_type(16))) _Float16 v16h;
typedef __attribute__((ext_vector_type(8)))  _Float16 v8h;
typedef __attribute__((ext_vector_type(8)))  float    v8f;
typedef __attribute__((ext_vector_type(4)))  unsigned int u32x4;
typedef __attribute__((ext_vector_type(8)))  int i32x8;
typedef __attribute__((ext_vector_type(4)))  int i32x4;

// LDS row stride for the A tile: 128 halves + 8 halves pad (16B) = 272B/row,
// so lane r's ds_load_b128 hits banks 4r..4r+3 (conflict-free across rows).
#define LDS_ROW 136

// ---------------- small device helpers ----------------
__device__ __forceinline__ float lrelu(float v) {
    return v > 0.0f ? v : NEG_SLOPE * v;
}

__device__ __forceinline__ void atomAddF(float* p, float v) {
    __hip_atomic_fetch_add(p, v, __ATOMIC_RELAXED, __HIP_MEMORY_SCOPE_AGENT);
}

// float atomic max via sign-split int/uint trick (works for all finite + -inf)
__device__ __forceinline__ void atomMaxF(float* p, float v) {
    if (v >= 0.0f) atomicMax((int*)p, __float_as_int(v));
    else           atomicMin((unsigned int*)p, (unsigned int)__float_as_int(v));
}

// edge endpoint fetch (self loops appended logically after E real edges)
__device__ __forceinline__ void edge_ends(const int* __restrict__ ei, int e,
                                          int& s, int& d) {
    if (e < N_EDGES) { s = ei[e]; d = ei[N_EDGES + e]; }
    else             { s = d = e - N_EDGES; }
}

// ---------------- utility kernels ----------------
__global__ void fill_f32(float* __restrict__ p, float v, int n) {
    int i = blockIdx.x * blockDim.x + threadIdx.x;
    if (i < n) p[i] = v;
}

__global__ void cvt_f32_f16(const float* __restrict__ x, _Float16* __restrict__ y, int n) {
    int i = blockIdx.x * blockDim.x + threadIdx.x;
    if (i < n) y[i] = (_Float16)x[i];
}

// Pack W1 [K=128, N=128] into per-wave B fragments:
// frag index = ((ntile*4 + kstep)*32 + lane), 16 halfs each.
// B (32x16 f16) layout: lanes 0-15 -> N=lane, elements j=0..15 -> K = kstep*32 + j
//                       lanes 16-31 -> N=lane-16, elements j -> K = kstep*32 + 16 + j
__global__ void pack_w1(const float* __restrict__ W1, _Float16* __restrict__ wfrag) {
    int t = blockIdx.x * blockDim.x + threadIdx.x;     // 0 .. 1023
    if (t >= 8 * 4 * 32) return;
    int lane  = t & 31;
    int ks    = (t >> 5) & 3;
    int ntile = t >> 7;
    int n     = ntile * 16 + (lane & 15);
    int kbase = ks * 32 + ((lane >> 4) ? 16 : 0);
    _Float16* dst = wfrag + (size_t)t * 16;
    #pragma unroll
    for (int j = 0; j < 16; ++j)
        dst[j] = (_Float16)W1[(size_t)(kbase + j) * F_IN + n];
}

// ---------------- conv1 GEMM: h1 = x @ W1  via TDM + LDS + WMMA ------------
// grid.x = 6250 M-tiles, 256 threads = 8 waves; wave w covers N-tile w.
// One TDM tensor_load_to_lds per WG pulls the 16x128 f16 A-tile into LDS
// (padded 16B per 256B row); all 8 waves build A fragments from LDS.
__global__ void __launch_bounds__(256)
gemm1_wmma(const _Float16* __restrict__ xh, const _Float16* __restrict__ wfrag,
           float* __restrict__ h1) {
    __shared__ _Float16 smem[16 * LDS_ROW];

    int wave = threadIdx.x >> 5;            // 0..7 -> N tile
    int lane = threadIdx.x & 31;
    int r    = lane & 15;
    int hi   = lane >> 4;

#if __has_builtin(__builtin_amdgcn_tensor_load_to_lds)
    if (wave == 0) {
        // ---- Tensor DMA descriptor (D#), groups 0 and 1 (2D tile) ----
        unsigned lds_base = (unsigned)(unsigned long long)(&smem[0]);
        unsigned long long ga =
            (unsigned long long)(xh + (size_t)blockIdx.x * 16 * F_IN);
        // group0: count=1 | lds_addr | global_addr[56:0] | type=2
        u32x4 g0 = { 1u,
                     lds_base,
                     (unsigned)(ga & 0xffffffffull),
                     (unsigned)((ga >> 32) & 0x01ffffffull) | (2u << 30) };
        // group1: data_size=1 (2B), pad_enable, pad_interval=5 (64 DWORDs),
        // pad_amount=3 (4 DWORDs = 16B); tensor_dim0=128 @bits[79:48],
        // tensor_dim1=16 @bits[111:80], tile_dim0=128 @bits[127:112],
        // tile_dim1=16 @bits[143:128], tensor_dim0_stride=128 @bits[207:160]
        i32x8 g1 = { (int)((1u << 16) | (1u << 20) | (5u << 22) | (3u << 25)),
                     (int)(128u << 16),
                     (int)(16u << 16),
                     (int)(128u << 16),
                     (int)16,
                     (int)128,
                     0, 0 };
        i32x4 gz4 = { 0, 0, 0, 0 };
        i32x8 gz8 = { 0, 0, 0, 0, 0, 0, 0, 0 };
        __builtin_amdgcn_tensor_load_to_lds(g0, g1, gz4, gz4, gz8, 0);
        __builtin_amdgcn_s_wait_tensorcnt((short)0);
    }
#else
    // cooperative fallback: 256 threads x 16B chunks into padded LDS rows
    {
        int t = threadIdx.x;
        int row = t >> 4, chunk = t & 15;
        *(v8h*)(&smem[row * LDS_ROW + chunk * 8]) =
            *(const v8h*)(xh + ((size_t)blockIdx.x * 16 + row) * F_IN + chunk * 8);
    }
#endif
    __syncthreads();

    v8f acc = {};
    #pragma unroll
    for (int ks = 0; ks < 4; ++ks) {
        // A fragment (16x32 f16) from LDS: lanes 0-15 rows, K runs {0..15};
        // lanes 16-31 same rows, K runs {16..31}
        union { v16h v; v8h h[2]; } a;
        int k0 = ks * 32 + hi * 16;
        a.h[0] = *(const v8h*)(&smem[r * LDS_ROW + k0]);
        a.h[1] = *(const v8h*)(&smem[r * LDS_ROW + k0 + 8]);
        // B fragment: pre-packed, one coalesced 32B load per lane
        v16h b = *(const v16h*)(wfrag + (size_t)(((wave * 4 + ks) * 32) + lane) * 16);
        acc = __builtin_amdgcn_wmma_f32_16x16x32_f16(
                  false, a.v, false, b, (short)0, acc, false, false);
    }
    // D layout: lane gives N=r; VGPR i gives M = i + 8*hi
    int col  = wave * 16 + r;
    int rowb = blockIdx.x * 16 + hi * 8;
    #pragma unroll
    for (int i = 0; i < 8; ++i)
        h1[(size_t)(rowb + i) * F_IN + col] = acc[i];
}

// ---------------- conv1 attention logits ----------------
__global__ void attn_logits1(const float* __restrict__ h1,
                             const float* __restrict__ a_src,
                             const float* __restrict__ a_dst,
                             float* __restrict__ als, float* __restrict__ ald) {
    int i = blockIdx.x * blockDim.x + threadIdx.x;  // n*8 + head
    if (i >= N_NODES * HEADS1) return;
    int hd = i & 7, n = i >> 3;
    const float* hp = h1 + (size_t)n * F_IN + hd * C1;
    float s = 0.f, d = 0.f;
    #pragma unroll
    for (int c = 0; c < C1; ++c) {
        float v = hp[c];
        s += v * a_src[hd * C1 + c];
        d += v * a_dst[hd * C1 + c];
    }
    als[i] = s; ald[i] = d;
}

// ---------------- conv1 edge passes ----------------
__global__ void edge_max1(const int* __restrict__ ei,
                          const float* __restrict__ als, const float* __restrict__ ald,
                          float* __restrict__ m) {
    int i = blockIdx.x * blockDim.x + threadIdx.x;   // edge*8 + head
    if (i >= ETOT * HEADS1) return;
    int hd = i & 7, e = i >> 3, s, d;
    edge_ends(ei, e, s, d);
    atomMaxF(&m[d * 8 + hd], lrelu(als[s * 8 + hd] + ald[d * 8 + hd]));
}

__global__ void edge_den1(const int* __restrict__ ei,
                          const float* __restrict__ als, const float* __restrict__ ald,
                          const float* __restrict__ m, float* __restrict__ dn) {
    int i = blockIdx.x * blockDim.x + threadIdx.x;
    if (i >= ETOT * HEADS1) return;
    int hd = i & 7, e = i >> 3, s, d;
    edge_ends(ei, e, s, d);
    float v = lrelu(als[s * 8 + hd] + ald[d * 8 + hd]);
    atomAddF(&dn[d * 8 + hd], __expf(v - m[d * 8 + hd]));
}

// 32 lanes per edge: lane handles 4 consecutive channels (one head per 4 lanes)
__global__ void edge_agg1(const int* __restrict__ ei,
                          const float* __restrict__ als, const float* __restrict__ ald,
                          const float* __restrict__ m, const float* __restrict__ dn,
                          const float* __restrict__ h1, float* __restrict__ agg) {
    int gid  = blockIdx.x * blockDim.x + threadIdx.x;
    int lane = gid & 31;
    int e    = gid >> 5;
    if (e >= ETOT) return;
    int s, d;
    edge_ends(ei, e, s, d);
    int c0 = lane * 4;
    int hd = c0 >> 4;
    float v     = lrelu(als[s * 8 + hd] + ald[d * 8 + hd]);
    float alpha = __expf(v - m[d * 8 + hd]) / (dn[d * 8 + hd] + 1e-16f);
    const float4 hv = *(const float4*)(h1 + (size_t)s * F_IN + c0);
    float* out = agg + (size_t)d * F_IN + c0;
    atomAddF(out + 0, hv.x * alpha);
    atomAddF(out + 1, hv.y * alpha);
    atomAddF(out + 2, hv.z * alpha);
    atomAddF(out + 3, hv.w * alpha);
}

// ---------------- bias + ELU ----------------
__global__ void bias_elu1(const float* __restrict__ agg, const float* __restrict__ b1,
                          float* __restrict__ act) {
    int i = blockIdx.x * blockDim.x + threadIdx.x;
    if (i >= N_NODES * F_IN) return;
    float v = agg[i] + b1[i & (F_IN - 1)];
    act[i] = v > 0.f ? v : __expf(v) - 1.f;
}

// ---------------- conv2 linear + attention logits (128 -> 2, 1 head) -------
__global__ void lin2(const float* __restrict__ act, const float* __restrict__ W2,
                     const float* __restrict__ a_src2, const float* __restrict__ a_dst2,
                     float* __restrict__ h2, float* __restrict__ als2,
                     float* __restrict__ ald2) {
    int n = blockIdx.x * blockDim.x + threadIdx.x;
    if (n >= N_NODES) return;
    const float* xp = act + (size_t)n * F_IN;
    float s0 = 0.f, s1 = 0.f;
    #pragma unroll 4
    for (int k = 0; k < F_IN; ++k) {
        float v = xp[k];
        s0 += v * W2[k * 2 + 0];
        s1 += v * W2[k * 2 + 1];
    }
    h2[n * 2 + 0] = s0;
    h2[n * 2 + 1] = s1;
    als2[n] = s0 * a_src2[0] + s1 * a_src2[1];
    ald2[n] = s0 * a_dst2[0] + s1 * a_dst2[1];
}

// ---------------- conv2 edge passes (head = 1) ----------------
__global__ void edge_max2(const int* __restrict__ ei, const float* __restrict__ als,
                          const float* __restrict__ ald, float* __restrict__ m) {
    int e = blockIdx.x * blockDim.x + threadIdx.x;
    if (e >= ETOT) return;
    int s, d;
    edge_ends(ei, e, s, d);
    atomMaxF(&m[d], lrelu(als[s] + ald[d]));
}

__global__ void edge_den2(const int* __restrict__ ei, const float* __restrict__ als,
                          const float* __restrict__ ald, const float* __restrict__ m,
                          float* __restrict__ dn) {
    int e = blockIdx.x * blockDim.x + threadIdx.x;
    if (e >= ETOT) return;
    int s, d;
    edge_ends(ei, e, s, d);
    atomAddF(&dn[d], __expf(lrelu(als[s] + ald[d]) - m[d]));
}

__global__ void edge_agg2(const int* __restrict__ ei, const float* __restrict__ als,
                          const float* __restrict__ ald, const float* __restrict__ m,
                          const float* __restrict__ dn, const float* __restrict__ h2,
                          float* __restrict__ agg) {
    int e = blockIdx.x * blockDim.x + threadIdx.x;
    if (e >= ETOT) return;
    int s, d;
    edge_ends(ei, e, s, d);
    float alpha = __expf(lrelu(als[s] + ald[d]) - m[d]) / (dn[d] + 1e-16f);
    atomAddF(&agg[d * 2 + 0], h2[s * 2 + 0] * alpha);
    atomAddF(&agg[d * 2 + 1], h2[s * 2 + 1] * alpha);
}

// ---------------- global mean pool + log_softmax ----------------
__global__ void pool_sum(const float* __restrict__ agg2, const float* __restrict__ b2,
                         const int* __restrict__ batch, float* __restrict__ psum,
                         float* __restrict__ pcnt) {
    int n = blockIdx.x * blockDim.x + threadIdx.x;
    if (n >= N_NODES) return;
    int g = batch[n];
    atomAddF(&psum[g * 2 + 0], agg2[n * 2 + 0] + b2[0]);
    atomAddF(&psum[g * 2 + 1], agg2[n * 2 + 1] + b2[1]);
    atomAddF(&pcnt[g], 1.0f);
}

__global__ void log_softmax_out(const float* __restrict__ psum,
                                const float* __restrict__ pcnt,
                                float* __restrict__ out) {
    int g = threadIdx.x;
    if (g >= NGRAPH) return;
    float c  = fmaxf(pcnt[g], 1.0f);
    float l0 = psum[g * 2 + 0] / c;
    float l1 = psum[g * 2 + 1] / c;
    float mx  = fmaxf(l0, l1);
    float lse = mx + logf(__expf(l0 - mx) + __expf(l1 - mx));
    out[g * 2 + 0] = l0 - lse;
    out[g * 2 + 1] = l1 - lse;
}

// ---------------- host side ----------------
static inline char* carve(char* base, size_t& off, size_t bytes) {
    off = (off + 255) & ~(size_t)255;
    char* p = base + off;
    off += bytes;
    return p;
}

extern "C" void kernel_launch(void* const* d_in, const int* in_sizes, int n_in,
                              void* d_out, int out_size, void* d_ws, size_t ws_size,
                              hipStream_t stream) {
    (void)in_sizes; (void)n_in; (void)out_size; (void)ws_size;

    const float* x       = (const float*)d_in[0];
    const int*   ei      = (const int*)d_in[1];   // [2, E] flattened; row0 src, row1 dst
    const int*   batch   = (const int*)d_in[2];
    const float* W1      = (const float*)d_in[3];
    const float* a_src1  = (const float*)d_in[4];
    const float* a_dst1  = (const float*)d_in[5];
    const float* b1      = (const float*)d_in[6];
    const float* W2      = (const float*)d_in[7];
    const float* a_src2  = (const float*)d_in[8];
    const float* a_dst2  = (const float*)d_in[9];
    const float* b2      = (const float*)d_in[10];
    float*       out     = (float*)d_out;

    // workspace carve
    size_t off = 0;
    char* ws = (char*)d_ws;
    _Float16* xh    = (_Float16*)carve(ws, off, (size_t)N_NODES * F_IN * 2);
    _Float16* wfrag = (_Float16*)carve(ws, off, (size_t)8 * 4 * 32 * 16 * 2);
    float* h1    = (float*)carve(ws, off, (size_t)N_NODES * F_IN * 4);  // also reused as act1
    float* als1  = (float*)carve(ws, off, (size_t)N_NODES * HEADS1 * 4);
    float* ald1  = (float*)carve(ws, off, (size_t)N_NODES * HEADS1 * 4);
    float* m1    = (float*)carve(ws, off, (size_t)N_NODES * HEADS1 * 4);
    float* dn1   = (float*)carve(ws, off, (size_t)N_NODES * HEADS1 * 4);
    float* agg1  = (float*)carve(ws, off, (size_t)N_NODES * F_IN * 4);
    float* h2    = (float*)carve(ws, off, (size_t)N_NODES * 2 * 4);
    float* als2  = (float*)carve(ws, off, (size_t)N_NODES * 4);
    float* ald2  = (float*)carve(ws, off, (size_t)N_NODES * 4);
    float* m2    = (float*)carve(ws, off, (size_t)N_NODES * 4);
    float* dn2   = (float*)carve(ws, off, (size_t)N_NODES * 4);
    float* agg2  = (float*)carve(ws, off, (size_t)N_NODES * 2 * 4);
    float* psum  = (float*)carve(ws, off, (size_t)NGRAPH * 2 * 4);
    float* pcnt  = (float*)carve(ws, off, (size_t)NGRAPH * 4);

    const int T = 256;
    const float NEG_INF = -__builtin_inff();
    #define GRID(n) (((n) + T - 1) / T)

    // init accumulators (ws is poisoned; must re-init every call)
    fill_f32<<<GRID(N_NODES * HEADS1), T, 0, stream>>>(m1, NEG_INF, N_NODES * HEADS1);
    fill_f32<<<GRID(N_NODES * HEADS1), T, 0, stream>>>(dn1, 0.f, N_NODES * HEADS1);
    fill_f32<<<GRID(N_NODES * F_IN),   T, 0, stream>>>(agg1, 0.f, N_NODES * F_IN);
    fill_f32<<<GRID(N_NODES), T, 0, stream>>>(m2, NEG_INF, N_NODES);
    fill_f32<<<GRID(N_NODES), T, 0, stream>>>(dn2, 0.f, N_NODES);
    fill_f32<<<GRID(N_NODES * 2), T, 0, stream>>>(agg2, 0.f, N_NODES * 2);
    fill_f32<<<1, T, 0, stream>>>(psum, 0.f, NGRAPH * 2);
    fill_f32<<<1, T, 0, stream>>>(pcnt, 0.f, NGRAPH);

    // conv1: TDM + WMMA GEMM
    cvt_f32_f16<<<GRID(N_NODES * F_IN), T, 0, stream>>>(x, xh, N_NODES * F_IN);
    pack_w1<<<4, T, 0, stream>>>(W1, wfrag);
    gemm1_wmma<<<N_NODES / 16, T, 0, stream>>>(xh, wfrag, h1);

    // conv1: attention + segment softmax + aggregation
    attn_logits1<<<GRID(N_NODES * HEADS1), T, 0, stream>>>(h1, a_src1, a_dst1, als1, ald1);
    edge_max1<<<GRID(ETOT * HEADS1), T, 0, stream>>>(ei, als1, ald1, m1);
    edge_den1<<<GRID(ETOT * HEADS1), T, 0, stream>>>(ei, als1, ald1, m1, dn1);
    edge_agg1<<<GRID(ETOT * 32), T, 0, stream>>>(ei, als1, ald1, m1, dn1, h1, agg1);

    // bias + ELU (act written into h1 buffer, whose contents are now dead)
    bias_elu1<<<GRID(N_NODES * F_IN), T, 0, stream>>>(agg1, b1, h1);

    // conv2
    lin2<<<GRID(N_NODES), T, 0, stream>>>(h1, W2, a_src2, a_dst2, h2, als2, ald2);
    edge_max2<<<GRID(ETOT), T, 0, stream>>>(ei, als2, ald2, m2);
    edge_den2<<<GRID(ETOT), T, 0, stream>>>(ei, als2, ald2, m2, dn2);
    edge_agg2<<<GRID(ETOT), T, 0, stream>>>(ei, als2, ald2, m2, dn2, h2, agg2);

    // pool + log_softmax
    pool_sum<<<GRID(N_NODES), T, 0, stream>>>(agg2, b2, batch, psum, pcnt);
    log_softmax_out<<<1, 64, 0, stream>>>(psum, pcnt, out);
    #undef GRID
}
